// UnitaryLSTMRegressor_8263517077564
// MI455X (gfx1250) — matrix operand order repaired and checked by reading, not compiled, not run on hardware
//
#include <hip/hip_runtime.h>
#include <cstdint>

// ---------------------------------------------------------------------------
// UnitaryLSTMRegressor for MI455X (gfx1250, wave32, WMMA f16 16x16x32)
//
//   prep_kernel : swizzle all weights into WMMA B-fragment layout (f16) in ws
//   scan_kernel : fused 2-layer LSTM scan + linear proj, 8 persistent WGs,
//                 recurrent state in LDS (A-fragment layout), async-to-LDS x
//                 staging, writes X into d_out
//   expm_kernel : per-wave 16x16 complex expm (scale + Taylor-Horner + masked
//                 squaring), 2 WMMAs per complex matmul, in-place on d_out
// ---------------------------------------------------------------------------

typedef __attribute__((ext_vector_type(16))) _Float16 v16h;
typedef __attribute__((ext_vector_type(8)))  float    v8f;

#define TT   1024
#define BB   128
#define DD   8
#define HH   256
#define NG   1024          // 4*H
#define BT   (BB*TT)

// workspace layout (uint32 units)
#define WC0_OFF 0                    // 64 tiles * 9 kb * 256 u32
#define WC1_OFF 147456               // 64 tiles * 16 kb * 256 u32
#define WL_OFF  409600               // 32 tiles * 8 kb * 256 u32
#define B0_OFF  475136               // 1024 f32
#define B1_OFF  476160               // 1024 f32

__device__ __forceinline__ v16h ld_frag(const uint32_t* p) {
  union U { uint4 q[2]; v16h v; } u;
  const uint4* p4 = (const uint4*)p;
  u.q[0] = p4[0];
  u.q[1] = p4[1];
  return u.v;
}

__device__ __forceinline__ v8f wmma16(v16h a, v16h b, v8f c) {
  return __builtin_amdgcn_wmma_f32_16x16x32_f16(false, a, false, b, (short)0, c,
                                                false, false);
}

__device__ __forceinline__ float sigf(float x) {
  return 1.0f / (1.0f + __expf(-x));
}

// ---------------------------------------------------------------------------
// prep: pack weights into wave32 B-fragment layout.
// B-frag, this code's contiguous k order: lane = (k%32>=16 ? 16:0) + n%16 ; vgpr v=(k%16)>>1,
// half e=k&1.  One u32 per (lane,v), two f16 halves.
// ---------------------------------------------------------------------------
__global__ void prep_kernel(const float* __restrict__ w_ih0,
                            const float* __restrict__ w_hh0,
                            const float* __restrict__ w_ih1,
                            const float* __restrict__ w_hh1,
                            const float* __restrict__ W_lin,
                            const float* __restrict__ b_ih0,
                            const float* __restrict__ b_hh0,
                            const float* __restrict__ b_ih1,
                            const float* __restrict__ b_hh1,
                            uint32_t* __restrict__ ws) {
  int tid = blockIdx.x * blockDim.x + threadIdx.x;

  // layer0 combined weight: K=288 (256 recurrent + 32-pad block holding D=8 input)
  if (tid < 288 * NG) {
    int k = tid >> 10, n = tid & 1023;
    float v;
    if (k < 256) v = w_hh0[n * 256 + k];
    else { int kk = k - 256; v = (kk < DD) ? w_ih0[n * DD + kk] : 0.0f; }
    int kb = k >> 5, kin = k & 31;
    int lane = ((kin >= 16) ? 16 : 0) + (n & 15);
    int vv = (kin & 15) >> 1, e = kin & 1;
    int tile = n >> 4;
    ((_Float16*)(ws + WC0_OFF + ((tile * 9 + kb) * 256 + lane * 8 + vv)))[e] =
        (_Float16)v;
  }
  // layer1 combined weight: K=512 ([h0 ; h1])
  if (tid < 512 * NG) {
    int k = tid >> 10, n = tid & 1023;
    float v = (k < 256) ? w_ih1[n * 256 + k] : w_hh1[n * 256 + (k - 256)];
    int kb = k >> 5, kin = k & 31;
    int lane = ((kin >= 16) ? 16 : 0) + (n & 15);
    int vv = (kin & 15) >> 1, e = kin & 1;
    int tile = n >> 4;
    ((_Float16*)(ws + WC1_OFF + ((tile * 16 + kb) * 256 + lane * 8 + vv)))[e] =
        (_Float16)v;
  }
  // linear head: (512, 256)
  if (tid < 256 * 512) {
    int k = tid >> 9, n = tid & 511;
    float v = W_lin[n * 256 + k];
    int kb = k >> 5, kin = k & 31;
    int lane = ((kin >= 16) ? 16 : 0) + (n & 15);
    int vv = (kin & 15) >> 1, e = kin & 1;
    int tile = n >> 4;
    ((_Float16*)(ws + WL_OFF + ((tile * 8 + kb) * 256 + lane * 8 + vv)))[e] =
        (_Float16)v;
  }
  if (tid < NG) {
    ((float*)(ws + B0_OFF))[tid] = b_ih0[tid] + b_hh0[tid];
    ((float*)(ws + B1_OFF))[tid] = b_ih1[tid] + b_hh1[tid];
  }
}

// ---------------------------------------------------------------------------
// scan: 8 workgroups x 1024 threads (32 waves). blockIdx = 16-row batch tile.
// Hidden states live in LDS in A-fragment layout:
//   lane = (k%32>=16 ? 16:0) + m ; v = (k%16)>>1 ; e = k&1  (contiguous k, this code's order)
// ---------------------------------------------------------------------------
__global__ __launch_bounds__(1024) void scan_kernel(
    const float* __restrict__ x, const uint32_t* __restrict__ ws,
    float* __restrict__ out) {
  __shared__ alignas(16) uint32_t hA0[9][32][8];   // h0 frags (kb 0..7) + x pad (kb 8)
  __shared__ alignas(16) uint32_t hA1[8][32][8];   // h1 frags
  __shared__ alignas(16) uint32_t thA[8][32][8];   // tanh(h1) frags
  __shared__ _Float16 g16[16][NG];                 // gate staging (32 KB)
  __shared__ alignas(16) float xs[16][8];          // async x staging (512 B)

  const uint32_t* wc0 = ws + WC0_OFF;
  const uint32_t* wc1 = ws + WC1_OFF;
  const uint32_t* wl  = ws + WL_OFF;
  const float* bias0 = (const float*)(ws + B0_OFF);
  const float* bias1 = (const float*)(ws + B1_OFF);

  const int tid = threadIdx.x;
  const int lane = tid & 31;
  const int w = tid >> 5;
  const int b0 = blockIdx.x * 16;
  const int m0 = (lane >> 4) * 8;
  const int nl = lane & 15;

  for (int i = tid; i < 9 * 32 * 8; i += 1024) ((uint32_t*)hA0)[i] = 0u;
  for (int i = tid; i < 8 * 32 * 8; i += 1024) {
    ((uint32_t*)hA1)[i] = 0u;
    ((uint32_t*)thA)[i] = 0u;
  }
  float c0r[4], c1r[4];
#pragma unroll
  for (int q = 0; q < 4; q++) { c0r[q] = 0.0f; c1r[q] = 0.0f; }
  __syncthreads();

  for (int t = 0; t < TT; t++) {
    // Re-define the big weight-stream pointers every iteration so LLVM cannot
    // hoist ~50 B-fragments (400+ VGPRs) out of the t-loop and spill them to
    // scratch; these streams are L2-resident (1.8 MB << 192 MB L2) and are
    // meant to be re-read each step.  The small projection stream `wl`
    // (8 frags = 64 VGPRs) is intentionally left hoistable.
    const uint32_t* wc0v = wc0;
    const uint32_t* wc1v = wc1;
    asm volatile("" : "+s"(wc0v), "+s"(wc1v));

    // (a0) async-copy x_t (16 rows x 8 f32 = 512B) into LDS via the gfx1250
    //      async-to-LDS datapath (ASYNCcnt-tracked).  Wave 0, full EXEC.
    if (w == 0) {
      int row = lane >> 1, part = lane & 1;
      const float* gp = &x[((size_t)(b0 + row) * TT + t) * DD + part * 4];
      uint32_t ldsoff = (uint32_t)(uintptr_t)(&xs[row][part * 4]);
      asm volatile("global_load_async_to_lds_b128 %0, %1, off"
                   :: "v"(ldsoff), "v"(gp)
                   : "memory");
      asm volatile("s_wait_asynccnt 0x0" ::: "memory");
    }
    if (tid < 16 && t + 1 < TT)
      __builtin_prefetch(&x[((size_t)(b0 + tid) * TT + (t + 1)) * DD], 0, 1);
    __syncthreads();

    // (a1) convert staged x into the K-pad block of layer0's A operand
    if (tid < 128) {
      int mm = tid >> 3, d = tid & 7;
      ((_Float16*)&hA0[8][mm][d >> 1])[d & 1] = (_Float16)xs[mm][d];
    }
    __syncthreads();

    // (b) layer0 gates: K = 288 (9 WMMA K-blocks), 64 N-tiles over 32 waves
#pragma unroll
    for (int it = 0; it < 2; it++) {
      int j = w * 2 + it;
      float bb = bias0[j * 16 + nl];
      v8f acc = {bb, bb, bb, bb, bb, bb, bb, bb};
#pragma unroll
      for (int kb = 0; kb < 9; kb++) {
        v16h a = ld_frag(&hA0[kb][lane][0]);
        v16h b = ld_frag(wc0v + ((j * 9 + kb) * 256 + lane * 8));
        acc = wmma16(a, b, acc);
      }
#pragma unroll
      for (int v = 0; v < 8; v++) g16[m0 + v][j * 16 + nl] = (_Float16)acc[v];
    }
    __syncthreads();

    // (c) layer0 cell/hidden update; write new h0 into A-frag layout
    {
      int base = tid * 4;
#pragma unroll
      for (int q = 0; q < 4; q++) {
        int idx = base + q, mm = idx >> 8, nn = idx & 255;
        float ig = sigf((float)g16[mm][nn]);
        float fg = sigf((float)g16[mm][nn + 256]);
        float gg = tanhf((float)g16[mm][nn + 512]);
        float og = sigf((float)g16[mm][nn + 768]);
        c0r[q] = fg * c0r[q] + ig * gg;
        float hh = og * tanhf(c0r[q]);
        int kin = nn & 31, hi = (kin >> 4) & 1, r2 = kin & 15;
        int ln = hi * 16 + mm, vv = r2 >> 1, ee = kin & 1;
        ((_Float16*)&hA0[nn >> 5][ln][vv])[ee] = (_Float16)hh;
      }
    }
    __syncthreads();

    // (d) layer1 gates: K = 512, A = [h0_t ; h1_{t-1}]
#pragma unroll
    for (int it = 0; it < 2; it++) {
      int j = w * 2 + it;
      float bb = bias1[j * 16 + nl];
      v8f acc = {bb, bb, bb, bb, bb, bb, bb, bb};
#pragma unroll
      for (int kb = 0; kb < 16; kb++) {
        const uint32_t* ap = (kb < 8) ? &hA0[kb][lane][0] : &hA1[kb - 8][lane][0];
        v16h a = ld_frag(ap);
        v16h b = ld_frag(wc1v + ((j * 16 + kb) * 256 + lane * 8));
        acc = wmma16(a, b, acc);
      }
#pragma unroll
      for (int v = 0; v < 8; v++) g16[m0 + v][j * 16 + nl] = (_Float16)acc[v];
    }
    __syncthreads();

    // (e) layer1 update; write h1 and tanh(h1) frags
    {
      int base = tid * 4;
#pragma unroll
      for (int q = 0; q < 4; q++) {
        int idx = base + q, mm = idx >> 8, nn = idx & 255;
        float ig = sigf((float)g16[mm][nn]);
        float fg = sigf((float)g16[mm][nn + 256]);
        float gg = tanhf((float)g16[mm][nn + 512]);
        float og = sigf((float)g16[mm][nn + 768]);
        c1r[q] = fg * c1r[q] + ig * gg;
        float hh = og * tanhf(c1r[q]);
        float th = tanhf(hh);
        int kin = nn & 31, hi = (kin >> 4) & 1, r2 = kin & 15;
        int ln = hi * 16 + mm, vv = r2 >> 1, ee = kin & 1;
        ((_Float16*)&hA1[nn >> 5][ln][vv])[ee] = (_Float16)hh;
        ((_Float16*)&thA[nn >> 5][ln][vv])[ee] = (_Float16)th;
      }
    }
    __syncthreads();

    // (f) projection: X[b,t,:512] = tanh(h1) @ W_lin^T  (one N-tile per wave)
    {
      int j = w;
      v8f acc = {0.f, 0.f, 0.f, 0.f, 0.f, 0.f, 0.f, 0.f};
#pragma unroll
      for (int kb = 0; kb < 8; kb++) {
        v16h a = ld_frag(&thA[kb][lane][0]);
        v16h b = ld_frag(wl + ((j * 8 + kb) * 256 + lane * 8));
        acc = wmma16(a, b, acc);
      }
#pragma unroll
      for (int v = 0; v < 8; v++) {
        out[((size_t)(b0 + m0 + v) * TT + t) * 512 + (j * 16 + nl)] = acc[v];
      }
    }
    __syncthreads();
  }
}

// ---------------------------------------------------------------------------
// expm: one wave per 16x16 complex matrix, in-place over d_out.
// Complex matmul = 2 WMMAs: [Ar|Ai](K=32) x [Mr;-Mi] and [Ar|Ai] x [Mi;Mr].
// ---------------------------------------------------------------------------
#define TAYLOR_N 12
#define MAXSQ    8

__global__ __launch_bounds__(256) void expm_kernel(float* __restrict__ out) {
  const int wv = threadIdx.x >> 5;
  const int lane = threadIdx.x & 31;
  const int mid = blockIdx.x * 8 + wv;          // grid sized so mid < BT exactly
  float* p = out + (size_t)mid * 512;

  __shared__ float sR[8][16][17];
  __shared__ float sI[8][16][17];
  float (*Rr)[17] = sR[wv];
  float (*Ri)[17] = sI[wv];

  // load X (real block then imag block)
#pragma unroll
  for (int q = 0; q < 8; q++) {
    int idx = lane + q * 32;
    Rr[idx >> 4][idx & 15] = p[idx];
    Ri[idx >> 4][idx & 15] = p[256 + idx];
  }
  __syncthreads();

  // A = (X - X^H)/2  (skew-Hermitian)
  float ar[8], ai[8];
#pragma unroll
  for (int q = 0; q < 8; q++) {
    int idx = lane + q * 32, r = idx >> 4, c = idx & 15;
    ar[q] = 0.5f * (Rr[r][c] - Rr[c][r]);
    ai[q] = 0.5f * (Ri[r][c] + Ri[c][r]);
  }
  __syncthreads();
#pragma unroll
  for (int q = 0; q < 8; q++) {
    int idx = lane + q * 32, r = idx >> 4, c = idx & 15;
    Rr[r][c] = ar[q];
    Ri[r][c] = ai[q];
  }
  __syncthreads();

  // induced 1-norm -> scaling exponent s
  float cs = 0.0f;
  {
    int cl = lane & 15;
#pragma unroll
    for (int r = 0; r < 16; r++)
      cs += sqrtf(Rr[r][cl] * Rr[r][cl] + Ri[r][cl] * Ri[r][cl]);
#pragma unroll
    for (int off = 1; off < 16; off <<= 1)
      cs = fmaxf(cs, __shfl_xor(cs, off, 32));
  }
  int s = (cs > 1.0f) ? (int)ceilf(log2f(cs)) : 0;
  if (s < 0) s = 0;
  if (s > MAXSQ) s = MAXSQ;
  float scl = exp2f(-(float)s);

  // fixed A operand frags: [Ar | Ai] along K=32
  v16h Acat;
  {
    int row = lane & 15, kh = lane >> 4;
#pragma unroll
    for (int v = 0; v < 8; v++)
#pragma unroll
      for (int e = 0; e < 2; e++) {
        int K = kh * 16 + v * 2 + e;
        float val = (K < 16) ? scl * Rr[row][K] : scl * Ri[row][K - 16];
        Acat[v * 2 + e] = (_Float16)val;
      }
  }

  const int nn = lane & 15;
  const int m0 = (lane >> 4) * 8;
  v8f Mr, Mi;
#pragma unroll
  for (int v = 0; v < 8; v++) {
    Mr[v] = (m0 + v == nn) ? 1.0f : 0.0f;
    Mi[v] = 0.0f;
  }

  // Taylor-Horner: M = I + A*M/k, k = N..1
  for (int k = TAYLOR_N; k >= 1; k--) {
    __syncthreads();
#pragma unroll
    for (int v = 0; v < 8; v++) { Rr[m0 + v][nn] = Mr[v]; Ri[m0 + v][nn] = Mi[v]; }
    __syncthreads();
    v16h b1, b2;
    {
      int kr = (lane >> 4) * 16;
#pragma unroll
      for (int v = 0; v < 8; v++)
#pragma unroll
        for (int e = 0; e < 2; e++) {
          int K = kr + v * 2 + e;
          float x1 = (K < 16) ? Rr[K][nn] : -Ri[K - 16][nn];
          float x2 = (K < 16) ? Ri[K][nn] : Rr[K - 16][nn];
          b1[v * 2 + e] = (_Float16)x1;
          b2[v * 2 + e] = (_Float16)x2;
        }
    }
    v8f Pr = {0.f, 0.f, 0.f, 0.f, 0.f, 0.f, 0.f, 0.f};
    v8f Pi = {0.f, 0.f, 0.f, 0.f, 0.f, 0.f, 0.f, 0.f};
    Pr = wmma16(Acat, b1, Pr);
    Pi = wmma16(Acat, b2, Pi);
    float inv = 1.0f / (float)k;
#pragma unroll
    for (int v = 0; v < 8; v++) {
      Mr[v] = ((m0 + v == nn) ? 1.0f : 0.0f) + Pr[v] * inv;
      Mi[v] = Pi[v] * inv;
    }
  }

  // masked repeated squaring (uniform trip count -> barriers are legal)
  for (int it = 0; it < MAXSQ; it++) {
    __syncthreads();
#pragma unroll
    for (int v = 0; v < 8; v++) { Rr[m0 + v][nn] = Mr[v]; Ri[m0 + v][nn] = Mi[v]; }
    __syncthreads();
    v16h Ma, b1, b2;
    {
      int row = lane & 15, kh = lane >> 4;
#pragma unroll
      for (int v = 0; v < 8; v++)
#pragma unroll
        for (int e = 0; e < 2; e++) {
          int K = kh * 16 + v * 2 + e;
          float val = (K < 16) ? Rr[row][K] : Ri[row][K - 16];
          Ma[v * 2 + e] = (_Float16)val;
        }
      int kr = (lane >> 4) * 16;
#pragma unroll
      for (int v = 0; v < 8; v++)
#pragma unroll
        for (int e = 0; e < 2; e++) {
          int K = kr + v * 2 + e;
          float x1 = (K < 16) ? Rr[K][nn] : -Ri[K - 16][nn];
          float x2 = (K < 16) ? Ri[K][nn] : Rr[K - 16][nn];
          b1[v * 2 + e] = (_Float16)x1;
          b2[v * 2 + e] = (_Float16)x2;
        }
    }
    v8f Pr = {0.f, 0.f, 0.f, 0.f, 0.f, 0.f, 0.f, 0.f};
    v8f Pi = {0.f, 0.f, 0.f, 0.f, 0.f, 0.f, 0.f, 0.f};
    Pr = wmma16(Ma, b1, Pr);
    Pi = wmma16(Ma, b2, Pi);
    if (it < s) {
#pragma unroll
      for (int v = 0; v < 8; v++) { Mr[v] = Pr[v]; Mi[v] = Pi[v]; }
    }
  }

  // write interleaved complex64 result in place
#pragma unroll
  for (int v = 0; v < 8; v++) {
    int idx = (m0 + v) * 16 + nn;
    p[2 * idx]     = Mr[v];
    p[2 * idx + 1] = Mi[v];
  }
}

// ---------------------------------------------------------------------------
extern "C" void kernel_launch(void* const* d_in, const int* in_sizes, int n_in,
                              void* d_out, int out_size, void* d_ws,
                              size_t ws_size, hipStream_t stream) {
  const float* x     = (const float*)d_in[0];
  const float* w_ih0 = (const float*)d_in[1];
  const float* w_hh0 = (const float*)d_in[2];
  const float* b_ih0 = (const float*)d_in[3];
  const float* b_hh0 = (const float*)d_in[4];
  const float* w_ih1 = (const float*)d_in[5];
  const float* w_hh1 = (const float*)d_in[6];
  const float* b_ih1 = (const float*)d_in[7];
  const float* b_hh1 = (const float*)d_in[8];
  const float* W_lin = (const float*)d_in[9];
  (void)in_sizes; (void)n_in; (void)out_size; (void)ws_size;

  uint32_t* ws = (uint32_t*)d_ws;   // needs ~1.9 MB
  float* outp = (float*)d_out;

  prep_kernel<<<2048, 256, 0, stream>>>(w_ih0, w_hh0, w_ih1, w_hh1, W_lin,
                                        b_ih0, b_hh0, b_ih1, b_hh1, ws);
  scan_kernel<<<BB / 16, 1024, 0, stream>>>(x, ws, outp);
  expm_kernel<<<BT / 8, 256, 0, stream>>>(outp);
}
